// ScaledDotProductAttention_30674656428203
// MI455X (gfx1250) — compile-verified, matrix operand order checked by reading, not run on hardware
//
#include <hip/hip_runtime.h>

typedef __attribute__((ext_vector_type(16))) __bf16 v16bf;
typedef __attribute__((ext_vector_type(8)))  float  v8f;

#define B_    8
#define H_    8
#define S_    1024
#define D_    64
#define QTILE 16              // q rows per wave
#define WAVES 8               // waves per block (256 threads, wave32)
#define QBLK  (QTILE * WAVES) // 128 q rows per block
#define JT    32              // keys per iteration
#define NT    (S_ / JT)       // 32 key tiles

#define HEADS     (B_ * H_)
#define KV_ELEMS  ((size_t)HEADS * S_ * D_)   // 4,194,304 elements per tensor
#define MW_WORDS  ((size_t)B_ * NT * S_)      // packed mask words: [b][jt][qrow]

union FragB16 {
    v16bf v;
    uint4 u[2];
};

__device__ __forceinline__ void cvt8_bf16(float4 a, float4 b, __bf16* d) {
    d[0] = (__bf16)a.x; d[1] = (__bf16)a.y; d[2] = (__bf16)a.z; d[3] = (__bf16)a.w;
    d[4] = (__bf16)b.x; d[5] = (__bf16)b.y; d[6] = (__bf16)b.z; d[7] = (__bf16)b.w;
}

// Async global->LDS copies with per-lane addressing (ASYNCcnt-tracked).
// Inline asm per CDNA5 ISA (portable across both toolchains).
__device__ __forceinline__ void async_copy_b128(void* ldsDst, const void* gSrc) {
    unsigned           l = (unsigned)(size_t)ldsDst;   // LDS aperture: addr[31:0] = LDS byte address
    unsigned long long g = (unsigned long long)(size_t)gSrc;
    asm volatile("global_load_async_to_lds_b128 %0, %1, off"
                 :: "v"(l), "v"(g) : "memory");
}
__device__ __forceinline__ void async_copy_b64(void* ldsDst, const void* gSrc) {
    unsigned           l = (unsigned)(size_t)ldsDst;
    unsigned long long g = (unsigned long long)(size_t)gSrc;
    asm volatile("global_load_async_to_lds_b64 %0, %1, off"
                 :: "v"(l), "v"(g) : "memory");
}

// Butterfly reductions within a 16-lane DPP row (ROW_XMASK, pure VALU -- no LDS).
template <int MASK>
__device__ __forceinline__ float dpp_xmask(float x) {
    int v = __builtin_amdgcn_update_dpp(0, __builtin_bit_cast(int, x),
                                        0x160 + MASK, 0xf, 0xf, true);
    return __builtin_bit_cast(float, v);
}
__device__ __forceinline__ float rowmax16(float x) {
    x = fmaxf(x, dpp_xmask<1>(x));
    x = fmaxf(x, dpp_xmask<2>(x));
    x = fmaxf(x, dpp_xmask<4>(x));
    x = fmaxf(x, dpp_xmask<8>(x));
    return x;
}
__device__ __forceinline__ float rowsum16(float x) {
    x += dpp_xmask<1>(x);
    x += dpp_xmask<2>(x);
    x += dpp_xmask<4>(x);
    x += dpp_xmask<8>(x);
    return x;
}

// ---------------- Pre-pass A: K -> bf16 (same layout), V -> bf16 transposed [bh][d][s] ----
__global__ __launch_bounds__(256) void cvt_kv_kernel(
    const float* __restrict__ k, const float* __restrict__ v,
    __bf16* __restrict__ kbf, __bf16* __restrict__ vtbf) {

    const int gid = blockIdx.x * 256 + threadIdx.x;   // 524288 threads
    const int bhs = gid >> 3;                         // (bh*S + s)
    const int c8  = (gid & 7) * 8;                    // d chunk base
    const int bh  = bhs >> 10;
    const int s   = bhs & (S_ - 1);
    const size_t base = (size_t)bhs * D_ + c8;

    float4 f0 = *(const float4*)(k + base);
    float4 f1 = *(const float4*)(k + base + 4);
    union { __bf16 h[8]; uint4 u; } pk;
    cvt8_bf16(f0, f1, pk.h);
    *(uint4*)(kbf + base) = pk.u;

    float4 g0 = *(const float4*)(v + base);
    float4 g1 = *(const float4*)(v + base + 4);
    __bf16 tv[8];
    cvt8_bf16(g0, g1, tv);
    __bf16* vt = vtbf + (size_t)bh * S_ * D_;
    #pragma unroll
    for (int i = 0; i < 8; ++i) vt[(size_t)(c8 + i) * S_ + s] = tv[i];
}

// ---------------- Pre-pass B: bit-pack mask to [b][jt][qrow] uint32 ----------------
__global__ __launch_bounds__(256) void pack_mask_kernel(
    const unsigned char* __restrict__ mask, unsigned* __restrict__ mbits) {

    const int gid  = blockIdx.x * 256 + threadIdx.x;  // 262144 words
    const int qrow = gid & (S_ - 1);
    const int bjt  = gid >> 10;                       // b*NT + jt
    const int b    = bjt >> 5;
    const int jt   = bjt & (NT - 1);

    const unsigned char* src = mask + ((size_t)b * S_ + qrow) * S_ + jt * JT;
    uint4 m0 = *(const uint4*)(src);
    uint4 m1 = *(const uint4*)(src + 16);
    const unsigned char* p0 = (const unsigned char*)&m0;
    const unsigned char* p1 = (const unsigned char*)&m1;
    unsigned w = 0;
    #pragma unroll
    for (int i = 0; i < 16; ++i) w |= (p0[i] ? 1u : 0u) << i;
    #pragma unroll
    for (int i = 0; i < 16; ++i) w |= (p1[i] ? 1u : 0u) << (16 + i);
    mbits[gid] = w;
}

// ---------------- Flash attention forward ----------------
__global__ __launch_bounds__(256) void fa_fwd_kernel(
    const float* __restrict__ q, const __bf16* __restrict__ kbf,
    const __bf16* __restrict__ vtbf, const unsigned* __restrict__ mbits,
    float* __restrict__ out) {

    __shared__ __align__(16) __bf16   ldsK[2][JT * D_];    // [s][d] row-major bf16
    __shared__ __align__(16) __bf16   ldsVt[2][D_ * JT];   // [d][s] transposed bf16
    __shared__ __align__(16) unsigned ldsMW[2][QBLK];      // packed mask word per qrow
    __shared__ __align__(16) __bf16   ldsP[WAVES * QTILE * JT];

    const int tid  = threadIdx.x;
    const int w    = tid >> 5;
    const int lane = tid & 31;
    const int half = lane >> 4;
    const int n    = lane & 15;

    const int bh = blockIdx.x >> 3;
    const int qc = blockIdx.x & 7;
    const int b  = bh >> 3;                 // H_ == 8

    const size_t headOff = (size_t)bh * S_ * D_;
    const float*  qh  = q + headOff;
    const __bf16* khb = kbf + headOff;
    const __bf16* vth = vtbf + headOff;
    float*        oh  = out + headOff;
    const unsigned* mbB = mbits + (size_t)b * NT * S_;

    const int qcbase = qc * QBLK;
    const int qbase  = qcbase + w * QTILE;

    // Stage one tile: K (4KB linear), V^T (64 x 64B rows), packed mask (128 words).
    // Uniform 3 async ops per thread per tile (mask copy redundantly covered 4x).
    auto issueTile = [&](int jt, int buf) {
        const int jb = jt * JT;
        async_copy_b128(&ldsK[buf][tid * 8], khb + (size_t)jb * D_ + tid * 8);
        const int d  = tid >> 2;
        const int ch = (tid & 3) * 8;
        async_copy_b128(&ldsVt[buf][d * JT + ch], vth + (size_t)d * S_ + jb + ch);
        const int mt = (tid & 63) * 2;
        async_copy_b64(&ldsMW[buf][mt], mbB + (size_t)jt * S_ + qcbase + mt);
    };

    // ---- Load Q tile once: A-layout bf16 fragments, contraction d split 2x32 ----
    v16bf qa[2];
    {
        const int qrow = qbase + n;
        #pragma unroll
        for (int kk = 0; kk < 2; ++kk) {
            #pragma unroll
            for (int g = 0; g < 2; ++g) {
                const float* src = qh + (size_t)qrow * D_ + kk * 32 + g * 16 + half * 8;
                float4 f0 = *(const float4*)(src);
                float4 f1 = *(const float4*)(src + 4);
                __bf16 t[8];
                cvt8_bf16(f0, f1, t);
                #pragma unroll
                for (int i = 0; i < 8; ++i) qa[kk][g * 8 + i] = t[i];
            }
        }
    }

    v8f o[4] = {};
    float m_run[8], l_run[8];
    #pragma unroll
    for (int r = 0; r < 8; ++r) { m_run[r] = -3.0e38f; l_run[r] = 0.0f; }

    __bf16* pd = &ldsP[w * (QTILE * JT)];

    issueTile(0, 0);   // prologue

    for (int jt = 0; jt < NT; ++jt) {
        const int cur = jt & 1;
        if (jt) __syncthreads();            // all waves done reading buf[cur^1]

        if (jt + 1 < NT) {
            issueTile(jt + 1, cur ^ 1);     // overlap next tile with this compute
            asm volatile("s_wait_asynccnt 3" ::: "memory");  // tile jt landed (in-order)
        } else {
            asm volatile("s_wait_asynccnt 0" ::: "memory");
        }
        __syncthreads();                    // tile jt visible to every wave

        // ---- S = Q K^T : two 16-col tiles, contraction 64 = 2 x K32 ----
        v8f s0 = {}, s1 = {};
        #pragma unroll
        for (int kk = 0; kk < 2; ++kk) {
            FragB16 kb0, kb1;
            const __bf16* r0 = &ldsK[cur][(0 + n) * D_ + kk * 32 + half * 8];
            const __bf16* r1 = &ldsK[cur][(16 + n) * D_ + kk * 32 + half * 8];
            kb0.u[0] = *(const uint4*)(r0);
            kb0.u[1] = *(const uint4*)(r0 + 16);
            kb1.u[0] = *(const uint4*)(r1);
            kb1.u[1] = *(const uint4*)(r1 + 16);
            s0 = __builtin_amdgcn_wmma_f32_16x16x32_bf16(false, qa[kk], false, kb0.v,
                                                         (short)0, s0, false, false);
            s1 = __builtin_amdgcn_wmma_f32_16x16x32_bf16(false, qa[kk], false, kb1.v,
                                                         (short)0, s1, false, false);
        }

        // ---- Online softmax over this 32-key tile ----
        #pragma unroll
        for (int r = 0; r < 8; ++r) {
            const unsigned mw = ldsMW[cur][w * QTILE + r + 8 * half];  // LDS broadcast
            float a0 = s0[r] * 0.125f;
            float a1 = s1[r] * 0.125f;
            a0 = ((mw >> n) & 1u)        ? a0 : -3.0e38f;
            a1 = ((mw >> (n + 16)) & 1u) ? a1 : -3.0e38f;

            const float t  = rowmax16(fmaxf(a0, a1));
            const float mn = fmaxf(m_run[r], t);
            const float alpha = __expf(m_run[r] - mn);
            m_run[r] = mn;

            const float e0 = __expf(a0 - mn);
            const float e1 = __expf(a1 - mn);
            const float rs = rowsum16(e0 + e1);
            l_run[r] = l_run[r] * alpha + rs;

            o[0][r] *= alpha; o[1][r] *= alpha; o[2][r] *= alpha; o[3][r] *= alpha;

            const int row = r + 8 * half;
            pd[row * JT + n]      = (__bf16)e0;
            pd[row * JT + 16 + n] = (__bf16)e1;
        }

        // ---- P as A-operand (per-wave scratch; DS ops are in-order per wave) ----
        FragB16 pa;
        pa.u[0] = *(const uint4*)(&pd[n * JT + half * 8]);
        pa.u[1] = *(const uint4*)(&pd[n * JT + 16 + half * 8]);

        // ---- O += P V : 4 output N-tiles of 16 cols ----
        #pragma unroll
        for (int nd = 0; nd < 4; ++nd) {
            FragB16 vb;
            const __bf16* vr = &ldsVt[cur][(nd * 16 + n) * JT + half * 8];
            vb.u[0] = *(const uint4*)(vr);
            vb.u[1] = *(const uint4*)(vr + 16);
            o[nd] = __builtin_amdgcn_wmma_f32_16x16x32_bf16(false, pa.v, false, vb.v,
                                                            (short)0, o[nd], false, false);
        }
    }

    // ---- Normalize and store ----
    #pragma unroll
    for (int r = 0; r < 8; ++r) {
        const float inv = 1.0f / l_run[r];
        float* dst = oh + (size_t)(qbase + r + 8 * half) * D_;
        #pragma unroll
        for (int nd = 0; nd < 4; ++nd) dst[nd * 16 + n] = o[nd][r] * inv;
    }
}

extern "C" void kernel_launch(void* const* d_in, const int* in_sizes, int n_in,
                              void* d_out, int out_size, void* d_ws, size_t ws_size,
                              hipStream_t stream) {
    const float* q = (const float*)d_in[0];
    const float* k = (const float*)d_in[1];
    const float* v = (const float*)d_in[2];
    const unsigned char* mask = (const unsigned char*)d_in[3];
    float* out = (float*)d_out;

    __bf16*   kbf   = (__bf16*)d_ws;               // 8.4 MB
    __bf16*   vtbf  = kbf + KV_ELEMS;              // 8.4 MB, transposed per head
    unsigned* mbits = (unsigned*)(vtbf + KV_ELEMS); // 1 MB packed mask

    cvt_kv_kernel<<<(int)(KV_ELEMS / 8 / 256), 256, 0, stream>>>(k, v, kbf, vtbf);
    pack_mask_kernel<<<(int)(MW_WORDS / 256), 256, 0, stream>>>(mask, mbits);

    // Flash attention: 64 heads x 8 q-chunks = 512 blocks of 8 wave32 waves
    fa_fwd_kernel<<<HEADS * (S_ / QBLK), 256, 0, stream>>>(q, kbf, vtbf, mbits, out);
}
